// DisentangledSelfAttention_2465311228233
// MI455X (gfx1250) — compile-verified, matrix operand order checked by reading, not run on hardware
//
#include <hip/hip_runtime.h>
#include <hip/hip_bf16.h>

// ---------------------------------------------------------------------------
// DeBERTa disentangled self-attention for MI455X (gfx1250), wave32 + WMMA.
//
// Identity used:  scores[i,j] = (ql[i].kl[j] + ql[i].pos_k[p] + kl[j].pos_q[p])
//                 / sqrt(192),   p = clip(i-j+512, 0, 1023)
// For a 16x16 (i,j) tile p lies in a 31-wide window -> two extra 16x32 WMMA
// tiles per score tile + LDS diagonal gather. Clip == clamped row load.
//
// ws layout (bf16): ql[B,H,S,hd] | kl[B,H,S,hd] | vlT[B,H,hd,S] |
//                   pos_k[H,2S,hd] | pos_q[H,2S,hd] | Wt[3][768][768] (~44 MB)
// ---------------------------------------------------------------------------

#define H_  12
#define D_  768
#define HD  64
#define S_  1024
#define B_  8
#define TWO_SPAN 1024

#define QL_ELEMS  (B_ * H_ * S_ * HD)      // 6291456
#define POS_ELEMS (H_ * TWO_SPAN * HD)     //  786432
#define W_ELEMS   (D_ * D_)                //  589824

typedef __attribute__((ext_vector_type(16))) __bf16 v16bf;
typedef __attribute__((ext_vector_type(8)))  float  v8f;

struct U4 { unsigned int x, y, z, w; };    // trivially-constructible 16B chunk
struct U2 { unsigned int x, y; };

union Frag {
    v16bf          v;
    unsigned short u[16];
    U4             q[2];
};

__device__ __forceinline__ unsigned short f2bf(float f) {
    unsigned int u = __float_as_uint(f);
    u += 0x7FFFu + ((u >> 16) & 1u);       // round-to-nearest-even
    return (unsigned short)(u >> 16);
}
__device__ __forceinline__ float bf2f(unsigned short h) {
    return __uint_as_float(((unsigned int)h) << 16);
}

// unpack/pack 8 bf16 <-> 8 f32 (for vectorized softmax)
__device__ __forceinline__ void unpack8(U4 c, float* f) {
    unsigned int w0 = c.x, w1 = c.y, w2 = c.z, w3 = c.w;
    f[0] = __uint_as_float(w0 << 16); f[1] = __uint_as_float(w0 & 0xFFFF0000u);
    f[2] = __uint_as_float(w1 << 16); f[3] = __uint_as_float(w1 & 0xFFFF0000u);
    f[4] = __uint_as_float(w2 << 16); f[5] = __uint_as_float(w2 & 0xFFFF0000u);
    f[6] = __uint_as_float(w3 << 16); f[7] = __uint_as_float(w3 & 0xFFFF0000u);
}
__device__ __forceinline__ U4 pack8(const float* f) {
    U4 c;
    c.x = (unsigned)f2bf(f[0]) | ((unsigned)f2bf(f[1]) << 16);
    c.y = (unsigned)f2bf(f[2]) | ((unsigned)f2bf(f[3]) << 16);
    c.z = (unsigned)f2bf(f[4]) | ((unsigned)f2bf(f[5]) << 16);
    c.w = (unsigned)f2bf(f[6]) | ((unsigned)f2bf(f[7]) << 16);
    return c;
}

// ---------------------------------------------------------------------------
// Kernel 0: transpose + convert weights once: Wt[z][n][k] = W_z[k][n] (bf16).
// Tiny (1.7M elems); weights then live in L2 for the whole projection pass.
// ---------------------------------------------------------------------------
__global__ __launch_bounds__(256)
void wtrans_kernel(const float* __restrict__ Wq, const float* __restrict__ Wk,
                   const float* __restrict__ Wv, unsigned short* __restrict__ wt)
{
    long idx = (long)blockIdx.x * 256 + threadIdx.x;
    if (idx >= 3L * W_ELEMS) return;
    int z   = (int)(idx / W_ELEMS);
    int rem = (int)(idx % W_ELEMS);
    int n = rem / D_, k = rem % D_;
    const float* W = (z == 0) ? Wq : (z == 1) ? Wk : Wv;
    wt[idx] = f2bf(W[(long)k * D_ + n]);
}

// ---------------------------------------------------------------------------
// Kernel 1: projections.  z: 0 q@Wq+bq->ql, 1 k@Wk+bk->kl, 2 v@Wv+bv->vlT,
//                         3 rel@Wk+bk->pos_k, 4 rel@Wq+bq->pos_q
// Block: 256 threads = 8 waves, each wave one 16x16 tile -> 32x64 block tile.
// A tile: float4 load -> bf16 pack -> ds_store_b64, double-buffered (1 barrier
// per K-step). B fragments: global_load_b128 from pre-transposed bf16 Wt.
// ---------------------------------------------------------------------------
__global__ __launch_bounds__(256)
void proj_kernel(const float* __restrict__ qin, const float* __restrict__ kin,
                 const float* __restrict__ vin, const float* __restrict__ rel,
                 const float* __restrict__ bq,  const float* __restrict__ bk,
                 const float* __restrict__ bv,
                 unsigned short* __restrict__ ws)
{
    const int z    = blockIdx.z;
    const int rows = (z < 3) ? (B_ * S_) : TWO_SPAN;
    const int r0   = blockIdx.x * 32;
    if (r0 >= rows) return;

    unsigned short* wql = ws;
    unsigned short* wkl = ws + (long)QL_ELEMS;
    unsigned short* wvl = ws + 2L * QL_ELEMS;
    unsigned short* wpk = ws + 3L * QL_ELEMS;
    unsigned short* wpq = wpk + (long)POS_ELEMS;
    const unsigned short* wtq = ws + 3L * QL_ELEMS + 2L * POS_ELEMS;
    const unsigned short* wtk = wtq + (long)W_ELEMS;
    const unsigned short* wtv = wtk + (long)W_ELEMS;

    const float* X; const unsigned short* Wt; const float* bias; unsigned short* dst;
    switch (z) {
        case 0:  X = qin; Wt = wtq; bias = bq; dst = wql; break;
        case 1:  X = kin; Wt = wtk; bias = bk; dst = wkl; break;
        case 2:  X = vin; Wt = wtv; bias = bv; dst = wvl; break;
        case 3:  X = rel; Wt = wtk; bias = bk; dst = wpk; break;
        default: X = rel; Wt = wtq; bias = bq; dst = wpq; break;
    }

    const int c0   = blockIdx.y * 64;
    const int tid  = threadIdx.x;
    const int lane = tid & 31;
    const int wid  = tid >> 5;        // 0..7
    const int wr   = wid >> 2;        // 0..1  (row tile)
    const int wc   = wid & 3;         // 0..3  (col tile)
    const int hi   = (lane >= 16);
    const int l16  = lane & 15;
    const int coff = hi ? 8 : 0;

    __shared__ __align__(16) unsigned short As[2][32][40];   // double-buffered A

    // per-thread staging coords: one float4 per K-step
    const int se = tid * 4;
    const int sr = se >> 5;           // 0..31
    const int scc = se & 31;          // 0,4,...,28
    const float* srow = &X[(long)(r0 + sr) * D_ + scc];

    // stage K-step 0 into buffer 0
    {
        float4 f = *(const float4*)srow;
        U2 p; p.x = (unsigned)f2bf(f.x) | ((unsigned)f2bf(f.y) << 16);
        p.y = (unsigned)f2bf(f.z) | ((unsigned)f2bf(f.w) << 16);
        *(U2*)&As[0][sr][scc] = p;
    }

    const unsigned short* wtrow = Wt + (long)(c0 + 16 * wc + l16) * D_;
    v8f acc = {};

    for (int it = 0; it < D_ / 32; ++it) {
        __syncthreads();
        const int cur = it & 1;
        const int kb  = it * 32;
        if (it + 1 < D_ / 32) {        // stage next K-step into other buffer
            float4 f = *(const float4*)(srow + kb + 32);
            U2 p; p.x = (unsigned)f2bf(f.x) | ((unsigned)f2bf(f.y) << 16);
            p.y = (unsigned)f2bf(f.z) | ((unsigned)f2bf(f.w) << 16);
            *(U2*)&As[1 - cur][sr][scc] = p;
        }
        Frag a, bf;
        const int m = 16 * wr + l16;
        a.q[0]  = *(const U4*)&As[cur][m][coff];        // ds_load_b128
        a.q[1]  = *(const U4*)&As[cur][m][16 + coff];
        bf.q[0] = *(const U4*)(wtrow + kb + coff);      // global_load_b128
        bf.q[1] = *(const U4*)(wtrow + kb + 16 + coff);
        acc = __builtin_amdgcn_wmma_f32_16x16x32_bf16(
                  false, a.v, false, bf.v, (short)0, acc, false, false);
    }

    const int   ncol = c0 + 16 * wc + l16;
    const float bb   = bias[ncol];
    const int   hh   = ncol >> 6;
    const int   dd   = ncol & 63;
#pragma unroll
    for (int vv = 0; vv < 8; ++vv) {
        int   Rr  = r0 + 16 * wr + vv + coff;
        float val = acc[vv] + bb;
        long  idx;
        if (z < 2) {               // ql/kl: [B,H,S,hd]
            int bidx = Rr >> 10, ss = Rr & 1023;
            idx = (((long)(bidx * H_ + hh)) * S_ + ss) * HD + dd;
        } else if (z == 2) {       // vlT: [B,H,hd,S]  (seq contiguous)
            int bidx = Rr >> 10, ss = Rr & 1023;
            idx = (((long)(bidx * H_ + hh)) * HD + dd) * S_ + ss;
        } else {                   // pos: [H,2S,hd]
            idx = (((long)hh) * TWO_SPAN + Rr) * HD + dd;
        }
        dst[idx] = f2bf(val);
    }
}

// ---------------------------------------------------------------------------
// Kernel 2: attention. Grid (64 i-blocks, H, B); 128 threads = 4 waves.
// Each wave handles 16 j-tiles (score row), then 16 output columns (PV).
// ---------------------------------------------------------------------------
__global__ __launch_bounds__(128)
void attn_kernel(const unsigned short* __restrict__ ws, float* __restrict__ out)
{
    const unsigned short* wql = ws;
    const unsigned short* wkl = ws + (long)QL_ELEMS;
    const unsigned short* wvl = ws + 2L * QL_ELEMS;
    const unsigned short* wpk = ws + 3L * QL_ELEMS;
    const unsigned short* wpq = wpk + (long)POS_ELEMS;

    const int i0   = blockIdx.x * 16;
    const int h    = blockIdx.y;
    const int b    = blockIdx.z;
    const int tid  = threadIdx.x;
    const int lane = tid & 31;
    const int wid  = tid >> 5;            // 0..3
    const int hi   = (lane >= 16);
    const int l16  = lane & 15;
    const int coff = hi ? 8 : 0;

    __shared__ __align__(16) unsigned short sc[16][1032];  // scores->probs (bf16)
    __shared__ float at_s[4][16][33];                      // per-wave c2p window
    __shared__ float bt_s[4][16][33];                      // per-wave p2c window
    __shared__ float red[16][8];                           // softmax reductions

    const long bh = (long)b * H_ + h;
    const unsigned short* qbase  = wql + bh * S_ * HD;
    const unsigned short* kbase  = wkl + bh * S_ * HD;
    const unsigned short* vbaseT = wvl + bh * HD * S_;     // [hd][S]
    const unsigned short* pkb    = wpk + (long)h * TWO_SPAN * HD;
    const unsigned short* pqb    = wpq + (long)h * TWO_SPAN * HD;

    // ql A-fragments for K windows 0 and 32 (row = query i0+l16, hd contiguous)
    Frag qf[2];
    {
        const unsigned short* qrow = qbase + (long)(i0 + l16) * HD;
#pragma unroll
        for (int t = 0; t < 2; ++t) {
            int kb = 32 * t;
            qf[t].q[0] = *(const U4*)(qrow + kb + coff);
            qf[t].q[1] = *(const U4*)(qrow + kb + 16 + coff);
        }
    }

    const float inv_scale = 0.07216878364870323f;  // 1/sqrt(hd*3)

    for (int jt = 0; jt < 16; ++jt) {
        const int j0 = (wid * 16 + jt) * 16;

        if (jt < 15)   // warm near cache for the next kl tile (global_prefetch_b8)
            __builtin_prefetch(kbase + (long)(j0 + 16 + l16) * HD, 0, 0);

        // kl fragment: serves as B-operand for QK^T and A-operand for p2c
        Frag kf[2];
        const unsigned short* krow = kbase + (long)(j0 + l16) * HD;
#pragma unroll
        for (int t = 0; t < 2; ++t) {
            int kb = 32 * t;
            kf[t].q[0] = *(const U4*)(krow + kb + coff);
            kf[t].q[1] = *(const U4*)(krow + kb + 16 + coff);
        }

        // content-content
        v8f accS = {};
        accS = __builtin_amdgcn_wmma_f32_16x16x32_bf16(
                   false, qf[0].v, false, kf[0].v, (short)0, accS, false, false);
        accS = __builtin_amdgcn_wmma_f32_16x16x32_bf16(
                   false, qf[1].v, false, kf[1].v, (short)0, accS, false, false);

        // 31-wide relative-position window; clamp == reference clip()
        const int pbase = i0 - j0 + 512 - 15;
#pragma unroll
        for (int t = 0; t < 2; ++t) {
            int pr = pbase + 16 * t + l16;
            pr = min(max(pr, 0), TWO_SPAN - 1);
            const unsigned short* pkrow = pkb + (long)pr * HD;
            const unsigned short* pqrow = pqb + (long)pr * HD;

            v8f accA = {};   // c2p: ql x pos_k^T  (M=di, N=dp)
            v8f accB = {};   // p2c: kl x pos_q^T  (M=dj, N=dp)
#pragma unroll
            for (int kk = 0; kk < 2; ++kk) {
                int kb = 32 * kk;
                Frag pf, pg;
                pf.q[0] = *(const U4*)(pkrow + kb + coff);
                pf.q[1] = *(const U4*)(pkrow + kb + 16 + coff);
                pg.q[0] = *(const U4*)(pqrow + kb + coff);
                pg.q[1] = *(const U4*)(pqrow + kb + 16 + coff);
                accA = __builtin_amdgcn_wmma_f32_16x16x32_bf16(
                           false, qf[kk].v, false, pf.v, (short)0, accA, false, false);
                accB = __builtin_amdgcn_wmma_f32_16x16x32_bf16(
                           false, kf[kk].v, false, pg.v, (short)0, accB, false, false);
            }
#pragma unroll
            for (int vv = 0; vv < 8; ++vv) {
                at_s[wid][vv + coff][16 * t + l16] = accA[vv];
                bt_s[wid][vv + coff][16 * t + l16] = accB[vv];
            }
        }

        // per-wave LDS RAW: DS ops are in-order, drain the counter
        asm volatile("s_wait_dscnt 0" ::: "memory");

        // diagonal gather: dp = di - dj + 15 in [0,30]
#pragma unroll
        for (int vv = 0; vv < 8; ++vv) {
            int di = vv + coff;
            int dj = l16;
            int dp = di - dj + 15;
            float s = (accS[vv] + at_s[wid][di][dp] + bt_s[wid][dj][dp]) * inv_scale;
            sc[di][j0 + dj] = f2bf(s);
        }
    }
    __syncthreads();

    // softmax over each of 16 rows (8 threads per row, b128-vectorized)
    {
        const int r  = tid >> 3;
        const int c8 = tid & 7;
        U4* rowp = (U4*)&sc[r][0];             // 128 valid 8-elem chunks

        float m = -1e30f;
        for (int i = c8; i < 128; i += 8) {
            float f[8]; unpack8(rowp[i], f);
#pragma unroll
            for (int j = 0; j < 8; ++j) m = fmaxf(m, f[j]);
        }
        red[r][c8] = m;
        __syncthreads();
        float mx = red[r][0];
#pragma unroll
        for (int i = 1; i < 8; ++i) mx = fmaxf(mx, red[r][i]);
        __syncthreads();

        float sum = 0.f;
        for (int i = c8; i < 128; i += 8) {
            float f[8]; unpack8(rowp[i], f);
#pragma unroll
            for (int j = 0; j < 8; ++j) { f[j] = __expf(f[j] - mx); sum += f[j]; }
            rowp[i] = pack8(f);
        }
        red[r][c8] = sum;
        __syncthreads();
        float tot = 0.f;
#pragma unroll
        for (int i = 0; i < 8; ++i) tot += red[r][i];
        float rinv = 1.0f / tot;
        for (int i = c8; i < 128; i += 8) {
            float f[8]; unpack8(rowp[i], f);
#pragma unroll
            for (int j = 0; j < 8; ++j) f[j] *= rinv;
            rowp[i] = pack8(f);
        }
    }
    __syncthreads();

    // ctx = probs (16x1024) x vlT (64x1024)^T; wave w -> columns 16w..16w+15
    {
        const int n0  = wid * 16;
        const int col = n0 + l16;
        const unsigned short* vrow = vbaseT + (long)col * S_;   // seq contiguous
        v8f acc = {};
#pragma unroll 4
        for (int kb = 0; kb < S_; kb += 32) {
            Frag a, bv;
            a.q[0]  = *(const U4*)&sc[l16][kb + coff];          // ds_load_b128
            a.q[1]  = *(const U4*)&sc[l16][kb + 16 + coff];
            bv.q[0] = *(const U4*)(vrow + kb + coff);           // global b128
            bv.q[1] = *(const U4*)(vrow + kb + 16 + coff);
            acc = __builtin_amdgcn_wmma_f32_16x16x32_bf16(
                      false, a.v, false, bv.v, (short)0, acc, false, false);
        }
#pragma unroll
        for (int vv = 0; vv < 8; ++vv) {
            int di = vv + coff;
            out[((long)(b * S_ + i0 + di)) * D_ + h * HD + col] = acc[vv];
        }
    }
}

// ---------------------------------------------------------------------------
extern "C" void kernel_launch(void* const* d_in, const int* in_sizes, int n_in,
                              void* d_out, int out_size, void* d_ws, size_t ws_size,
                              hipStream_t stream)
{
    const float* q   = (const float*)d_in[0];
    const float* k   = (const float*)d_in[1];
    const float* v   = (const float*)d_in[2];
    const float* rel = (const float*)d_in[3];
    const float* Wq  = (const float*)d_in[4];
    const float* bq  = (const float*)d_in[5];
    const float* Wk  = (const float*)d_in[6];
    const float* bk  = (const float*)d_in[7];
    const float* Wv  = (const float*)d_in[8];
    const float* bv  = (const float*)d_in[9];

    unsigned short* ws = (unsigned short*)d_ws;   // needs ~44.5 MB of d_ws
    float* out = (float*)d_out;

    unsigned short* wt = ws + 3L * QL_ELEMS + 2L * POS_ELEMS;
    long wt_total = 3L * W_ELEMS;
    dim3 g0((unsigned)((wt_total + 255) / 256)), b0(256);
    wtrans_kernel<<<g0, b0, 0, stream>>>(Wq, Wk, Wv, wt);

    dim3 g1(256, 12, 5), b1(256);
    proj_kernel<<<g1, b1, 0, stream>>>(q, k, v, rel, bq, bk, bv, ws);

    dim3 g2(S_ / 16, H_, B_), b2(128);
    attn_kernel<<<g2, b2, 0, stream>>>(ws, out);
}
